// Model_39676907886819
// MI455X (gfx1250) — matrix operand order repaired and checked by reading, not compiled, not run on hardware
//
#include <hip/hip_runtime.h>
#include <hip/hip_bf16.h>

typedef _Float16 h8   __attribute__((ext_vector_type(8)));
typedef _Float16 v16h __attribute__((ext_vector_type(16)));
typedef float    v8f  __attribute__((ext_vector_type(8)));

#define B_  2
#define H_  16
#define S_  2048
#define D_  64
#define BM  128     // query rows per block (8 waves x 16)
#define BN  64      // key rows per iteration
#define NBLK (S_ / BN)
#define LSTR 72     // LDS row stride in halves (144B: 16B-aligned rows, bank-skewed)
#define REGION (64 * LSTR)   // one 64-row staging region, in halves

#if __has_builtin(__builtin_amdgcn_exp2f)
#define EXP2(x) __builtin_amdgcn_exp2f(x)
#else
#define EXP2(x) exp2f(x)
#endif

// A-matrix 16x32 f16 fragment (ISA 7.12.2): row = lane%16, group g = lane/16,
// halves e<8 -> k = 32c + 8g + e ; e>=8 -> k = 32c + 8g + 16 + (e-8)
__device__ inline v16h load_a_frag(const _Float16* rowbase, int c, int g) {
    const _Float16* p = rowbase + c * 32 + g * 8;
    h8 lo = *(const h8*)p;
    h8 hi = *(const h8*)(p + 16);
    v16h a;
#pragma unroll
    for (int i = 0; i < 8; ++i) { a[i] = lo[i]; a[i + 8] = hi[i]; }
    return a;
}

// B-matrix 32x16 f16 fragment: column n = lane%16, halves e<8 -> k = 32c + 8g + e ; e>=8 -> k = 32c + 8g + 16 + (e-8)
__device__ inline v16h load_b_frag(const _Float16* colbase, int c, int g) {
    const _Float16* p = colbase + c * 32 + g * 8;
    h8 lo = *(const h8*)p;
    h8 hi = *(const h8*)(p + 16);
    v16h b;
#pragma unroll
    for (int i = 0; i < 8; ++i) { b[i] = lo[i]; b[i + 8] = hi[i]; }
    return b;
}

// ---- double-buffered K/V staging -------------------------------------------
// 256 4x4-subtile tasks cover a 64x64 block; each of 256 threads takes 1 task.
// K stored row-major f16, V stored transposed f16 (both with b64 stores).
__device__ inline void issue_loads(const float* kblk, const float* vblk, int tid,
                                   float4 kreg[4], float4 vreg[4]) {
    const int rg = tid >> 4, cq = tid & 15;
#pragma unroll
    for (int j = 0; j < 4; ++j) {
        kreg[j] = ((const float4*)kblk)[(4 * rg + j) * 16 + cq];
        vreg[j] = ((const float4*)vblk)[(4 * rg + j) * 16 + cq];
    }
}

__device__ inline void store_block(_Float16* Ksb, _Float16* Vtb, int tid,
                                   const float4 kreg[4], const float4 vreg[4]) {
    const int rg = tid >> 4, cq = tid & 15;
#pragma unroll
    for (int j = 0; j < 4; ++j) {          // K: row-major, 4 halves per b64
        const float* f = (const float*)&kreg[j];
        _Float16* d = &Ksb[(4 * rg + j) * LSTR + 4 * cq];
        d[0] = (_Float16)f[0]; d[1] = (_Float16)f[1];
        d[2] = (_Float16)f[2]; d[3] = (_Float16)f[3];
    }
#pragma unroll
    for (int i = 0; i < 4; ++i) {          // V: transposed 4x4, 4 halves per b64
        _Float16* d = &Vtb[(4 * cq + i) * LSTR + 4 * rg];
        d[0] = (_Float16)((const float*)&vreg[0])[i];
        d[1] = (_Float16)((const float*)&vreg[1])[i];
        d[2] = (_Float16)((const float*)&vreg[2])[i];
        d[3] = (_Float16)((const float*)&vreg[3])[i];
    }
}

__global__ __launch_bounds__(256) void fa_fwd_kernel(
    const float* __restrict__ q, const float* __restrict__ k,
    const float* __restrict__ v, const float* __restrict__ mask,
    float* __restrict__ out)
{
    // LDS (halves): Qs (2R) | Ks[2] (R each) | Vt[2] (R each) | Ps (2R) = 8R
    __shared__ _Float16 smem[8 * REGION];
    _Float16* Qs = smem;                       // regions 0-1 (128 rows)
    _Float16* Ps = smem + 6 * REGION;          // regions 6-7 (8 waves x 16 rows)

    const int tid  = threadIdx.x;
    const int wave = tid >> 5;
    const int lane = tid & 31;
    const int ln16 = lane & 15;
    const int g    = lane >> 4;

    const int bx   = blockIdx.x;
    const int mblk = bx & 15;            // S_/BM = 16
    const int h    = (bx >> 4) & 15;
    const int b    = bx >> 8;

    const long bh = (long)(b * H_ + h);
    const float* qb = q + (bh * S_ + (long)mblk * BM) * D_;
    const float* kb = k + bh * S_ * D_;
    const float* vb = v + bh * S_ * D_;
    const float* mb = mask + (long)b * S_ * S_ + (long)(mblk * BM) * S_;
    float* ob = out + (bh * S_ + (long)mblk * BM) * D_;

    const float scale = 0.125f;          // 1/sqrt(64), folded into Q

    // ---- stage Q (pre-scaled) to LDS as f16: 128x64 f32 = 2048 float4 ----
#pragma unroll
    for (int i = 0; i < 8; ++i) {
        int vi  = i * 256 + tid;
        int row = vi >> 4;
        int col = (vi & 15) << 2;
        float4 f = ((const float4*)qb)[vi];
        _Float16* d = &Qs[row * LSTR + col];
        d[0] = (_Float16)(f.x * scale); d[1] = (_Float16)(f.y * scale);
        d[2] = (_Float16)(f.z * scale); d[3] = (_Float16)(f.w * scale);
    }

    // ---- prologue: stage K/V block 0 into buffer 0 ----
    {
        float4 kreg[4], vreg[4];
        issue_loads(kb, vb, tid, kreg, vreg);
        store_block(smem + 2 * REGION, smem + 4 * REGION, tid, kreg, vreg);
    }
    __syncthreads();

    // per-wave Q A-fragments (rows wave*16 .. +15), two K=32 chunks
    v16h qa[2];
#pragma unroll
    for (int c = 0; c < 2; ++c)
        qa[c] = load_a_frag(&Qs[(wave * 16 + ln16) * LSTR], c, g);

    // flash state: rows r+8g per lane-half (C-layout row mapping)
    float mrow[8], lrow[8];
    v8f acc_o[4];
#pragma unroll
    for (int r = 0; r < 8; ++r) { mrow[r] = -INFINITY; lrow[r] = 0.0f; }
#pragma unroll
    for (int t = 0; t < 4; ++t) { v8f z = {}; acc_o[t] = z; }

    for (int nb = 0; nb < NBLK; ++nb) {
        const int cur = nb & 1, nxt = cur ^ 1;
        _Float16* Ks = smem + (2 + cur) * REGION;
        _Float16* Vt = smem + (4 + cur) * REGION;

        // ---- S = (Q/sqrtD) K^T : 4 tiles of 16x16, batch fragments per chunk ----
        v8f st[4];
#pragma unroll
        for (int c = 0; c < 2; ++c) {
            v16h bf[4];
#pragma unroll
            for (int t = 0; t < 4; ++t)
                bf[t] = load_b_frag(&Ks[(t * 16 + ln16) * LSTR], c, g);
#pragma unroll
            for (int t = 0; t < 4; ++t) {
                v8f cin;
                if (c == 0) { v8f z = {}; cin = z; } else cin = st[t];
                st[t] = __builtin_amdgcn_wmma_f32_16x16x32_f16(
                            false, qa[c], false, bf[t], (short)0, cin, false, false);
            }
        }

        // issue next block's global loads now: softmax + PV + barrier hide the
        // (L2-resident) latency, and the QK phase above ran at low VGPR pressure
        float4 kreg[4], vreg[4];
        if (nb + 1 < NBLK)
            issue_loads(kb + (long)(nb + 1) * BN * D_,
                        vb + (long)(nb + 1) * BN * D_, tid, kreg, vreg);

        // additive mask (broadcast over heads)
#pragma unroll
        for (int t = 0; t < 4; ++t)
#pragma unroll
            for (int r = 0; r < 8; ++r) {
                int m = wave * 16 + r + 8 * g;
                int n = nb * BN + t * 16 + ln16;
                st[t][r] += mb[(long)m * S_ + n];
            }

        // ---- online softmax over the 64 new columns ----
#pragma unroll
        for (int r = 0; r < 8; ++r) {
            float mx = fmaxf(fmaxf(st[0][r], st[1][r]), fmaxf(st[2][r], st[3][r]));
#pragma unroll
            for (int off = 1; off < 16; off <<= 1)
                mx = fmaxf(mx, __shfl_xor(mx, off, 16));
            float mnew = fmaxf(mrow[r], mx);
            float corr = EXP2((mrow[r] - mnew) * 1.44269504f);
            float rs = 0.0f;
#pragma unroll
            for (int t = 0; t < 4; ++t) {
                float p = EXP2((st[t][r] - mnew) * 1.44269504f);
                st[t][r] = p;
                rs += p;
            }
#pragma unroll
            for (int off = 1; off < 16; off <<= 1)
                rs += __shfl_xor(rs, off, 16);
            lrow[r] = lrow[r] * corr + rs;
            mrow[r] = mnew;
#pragma unroll
            for (int t = 0; t < 4; ++t) acc_o[t][r] *= corr;
        }

        // ---- P tile (f16) to per-wave LDS region (wave-private; LDS in-order) ----
        _Float16* Pw = Ps + wave * 16 * LSTR;
#pragma unroll
        for (int t = 0; t < 4; ++t)
#pragma unroll
            for (int r = 0; r < 8; ++r)
                Pw[(r + 8 * g) * LSTR + t * 16 + ln16] = (_Float16)st[t][r];

        v16h pa[2];
#pragma unroll
        for (int c = 0; c < 2; ++c)
            pa[c] = load_a_frag(&Pw[ln16 * LSTR], c, g);

        // ---- O += P V : 4 d-tiles, batch fragments per chunk ----
#pragma unroll
        for (int c = 0; c < 2; ++c) {
            v16h vf[4];
#pragma unroll
            for (int t = 0; t < 4; ++t)
                vf[t] = load_b_frag(&Vt[(t * 16 + ln16) * LSTR], c, g);
#pragma unroll
            for (int t = 0; t < 4; ++t)
                acc_o[t] = __builtin_amdgcn_wmma_f32_16x16x32_f16(
                            false, pa[c], false, vf[t], (short)0, acc_o[t], false, false);
        }

        // ---- pipeline: publish next block into the other LDS buffer ----
        __syncthreads();                  // all waves done reading buf[nxt] (prev iter)
        if (nb + 1 < NBLK)
            store_block(smem + (2 + nxt) * REGION, smem + (4 + nxt) * REGION,
                        tid, kreg, vreg);
        __syncthreads();                  // next block visible to all waves
    }

    // ---- normalize and store ----
#pragma unroll
    for (int r = 0; r < 8; ++r) {
        float inv = 1.0f / lrow[r];
        int m = wave * 16 + r + 8 * g;
#pragma unroll
        for (int t = 0; t < 4; ++t)
            ob[(long)m * D_ + t * 16 + ln16] = acc_o[t][r] * inv;
    }
}

extern "C" void kernel_launch(void* const* d_in, const int* in_sizes, int n_in,
                              void* d_out, int out_size, void* d_ws, size_t ws_size,
                              hipStream_t stream) {
    (void)in_sizes; (void)n_in; (void)out_size; (void)d_ws; (void)ws_size;
    const float* q    = (const float*)d_in[0];
    const float* k    = (const float*)d_in[1];
    const float* v    = (const float*)d_in[2];
    const float* mask = (const float*)d_in[3];
    float* out = (float*)d_out;
    dim3 grid(B_ * H_ * (S_ / BM));   // 2*16*16 = 512 blocks
    dim3 block(256);                  // 8 waves (wave32)
    hipLaunchKernelGGL(fa_fwd_kernel, grid, block, 0, stream, q, k, v, mask, out);
}
